// FastRCNNHead_59124519797117
// MI455X (gfx1250) — matrix operand order repaired and checked by reading, not compiled, not run on hardware
//
#include <hip/hip_runtime.h>
#include <hip/hip_bf16.h>

#define RESB   7
#define CCH    256
#define NROI   2048    // B*N = 4*512
#define FLATD  12544   // 256*7*7
#define FCD    1024
#define OUTD   93

// GEMM tiling
#define BM     128
#define BN     64
#define BK     32
#define KPAD   40      // LDS row pitch in u16: 80B/row -> 16B aligned, 16-way bank spread

typedef __attribute__((ext_vector_type(16))) __bf16    v16bf;
typedef __attribute__((ext_vector_type(8)))  float     v8f;
typedef __attribute__((ext_vector_type(4)))  unsigned  uint4v;

union FragBF { v16bf v; uint4v q[2]; };

__device__ __forceinline__ unsigned short f32_to_bf16_bits(float x) {
  unsigned u = __float_as_uint(x);
  u += 0x7FFFu + ((u >> 16) & 1u);   // round-to-nearest-even
  return (unsigned short)(u >> 16);
}

// ---------------------------------------------------------------------------
// Elementwise fp32 -> bf16 (weight conversion)
// ---------------------------------------------------------------------------
__global__ void __launch_bounds__(256) f32_to_bf16_kernel(
    const float* __restrict__ in, unsigned short* __restrict__ out, int n) {
  int i = blockIdx.x * 256 + threadIdx.x;
  if (i < n) out[i] = f32_to_bf16_bits(in[i]);
}

// ---------------------------------------------------------------------------
// ROI align (aligned=True semantics), 3-level FPN select, bf16 output
// one thread per (roi, channel, bin); feature maps are L2-resident (88 MB)
// ---------------------------------------------------------------------------
__global__ void __launch_bounds__(256) roi_align_kernel(
    const float* __restrict__ f3, const float* __restrict__ f4,
    const float* __restrict__ f5, const float* __restrict__ boxes,
    const int* __restrict__ anchor, unsigned short* __restrict__ pooled) {
  int idx = blockIdx.x * 256 + threadIdx.x;
  if (idx >= NROI * FLATD) return;
  int roi = idx / FLATD;
  int rem = idx - roi * FLATD;
  int c   = rem / 49;
  int bin = rem - c * 49;
  int ph  = bin / 7, pw = bin - ph * 7;
  int batch = roi >> 9;               // roi / 512

  int lvl = anchor[roi] / 3;
  lvl = lvl < 0 ? 0 : (lvl > 2 ? 2 : lvl);
  float scale; int H; const float* f;
  if (lvl == 0)      { scale = 0.125f;   H = 128; f = f3; }
  else if (lvl == 1) { scale = 0.0625f;  H = 64;  f = f4; }
  else               { scale = 0.03125f; H = 32;  f = f5; }
  const int W = H;
  const float* fmap = f + ((size_t)batch * CCH + c) * (size_t)H * W;

  float x1 = boxes[roi * 4 + 0] * scale - 0.5f;
  float y1 = boxes[roi * 4 + 1] * scale - 0.5f;
  float x2 = boxes[roi * 4 + 2] * scale - 0.5f;
  float y2 = boxes[roi * 4 + 3] * scale - 0.5f;
  float bh = (y2 - y1) * (1.0f / RESB);
  float bw = (x2 - x1) * (1.0f / RESB);

  float acc = 0.f;
  #pragma unroll
  for (int iy = 0; iy < 2; ++iy) {
    float y = y1 + bh * ((float)ph + (iy + 0.5f) * 0.5f);
    float my = (y >= -1.0f && y <= (float)H) ? 1.f : 0.f;
    float yc = fminf(fmaxf(y, 0.f), (float)(H - 1));
    int y0 = (int)floorf(yc); if (y0 > H - 1) y0 = H - 1;
    int y1i = (y0 + 1 > H - 1) ? H - 1 : y0 + 1;
    float ly = yc - (float)y0, hy = 1.f - ly;
    #pragma unroll
    for (int ix = 0; ix < 2; ++ix) {
      float x = x1 + bw * ((float)pw + (ix + 0.5f) * 0.5f);
      float mx = (x >= -1.0f && x <= (float)W) ? 1.f : 0.f;
      float xc = fminf(fmaxf(x, 0.f), (float)(W - 1));
      int x0 = (int)floorf(xc); if (x0 > W - 1) x0 = W - 1;
      int x1b = (x0 + 1 > W - 1) ? W - 1 : x0 + 1;
      float lx = xc - (float)x0, hx = 1.f - lx;
      float f00 = fmap[y0  * W + x0], f01 = fmap[y0  * W + x1b];
      float f10 = fmap[y1i * W + x0], f11 = fmap[y1i * W + x1b];
      float v = hy * (hx * f00 + lx * f01) + ly * (hx * f10 + lx * f11);
      acc += v * (my * mx);
    }
  }
  acc *= 0.25f;
  pooled[(size_t)roi * FLATD + rem] = f32_to_bf16_bits(acc);
}

// ---------------------------------------------------------------------------
// C = act(A[M,K] * B[N,K]^T + bias), bf16 in, f32 WMMA accumulate.
// 128x64 block tile, 8 waves (4x2), wave tile 32x32 = 2x2 WMMA tiles.
// Double-buffered LDS staged through registers. Per k-step:
//   barrier -> issue next-tile global b128s -> 8 ds_load_b128 + 4 WMMA
//   -> wait loads -> oob-select -> ds_store into alternate buffer.
// The global-load wait lands AFTER the WMMA block, so compute hides latency.
// ---------------------------------------------------------------------------
template <bool RELU, bool OUT_BF16>
__global__ void __launch_bounds__(256) gemm_bf16_wmma(
    const unsigned short* __restrict__ A,   // [M][K] bf16 bits
    const unsigned short* __restrict__ Bw,  // [N][K] bf16 bits (weight rows)
    const float* __restrict__ bias,         // [N]
    float* __restrict__ Cf,                 // [M][N]  (if !OUT_BF16)
    unsigned short* __restrict__ Cb,        // [M][N]  (if OUT_BF16)
    int M, int N, int K) {
  __shared__ alignas(16) unsigned short lA[2][BM * KPAD];
  __shared__ alignas(16) unsigned short lB[2][BN * KPAD];

  const int tid  = threadIdx.x;
  const int lane = tid & 31, wid = tid >> 5;
  const int wm = wid >> 1, wn = wid & 1;        // 4x2 wave grid, wave tile 32x32
  const int lo = lane & 15, hi = lane >> 4;
  const int rowBase = blockIdx.y * BM;
  const int colBase = blockIdx.x * BN;

  // staging coordinates: A has 512 16B chunks (2/thread), B has 256 (1/thread)
  const int sr = tid >> 2;                      // 0..63
  const int sq = (tid & 3) * 8;                 // u16 offset of 16B quarter

  // loop-invariant B-row guard: clamp the address, zero the value at store
  const int  nGlob  = colBase + sr;
  const int  nClamp = nGlob < N ? nGlob : N - 1;
  const bool bOob   = nGlob >= N;

  // advancing tile pointers (64 bytes per k-step)
  const unsigned short* pa0 = A  + (size_t)(rowBase + sr)      * K + sq;
  const unsigned short* pa1 = A  + (size_t)(rowBase + sr + 64) * K + sq;
  const unsigned short* pb  = Bw + (size_t)nClamp * K + sq;

  v8f acc[2][2];
  #pragma unroll
  for (int i = 0; i < 2; ++i)
    #pragma unroll
    for (int j = 0; j < 2; ++j)
      acc[i][j] = (v8f){0.f, 0.f, 0.f, 0.f, 0.f, 0.f, 0.f, 0.f};

  uint4v ra0, ra1, rb0;
  auto loadTiles = [&]() {
    ra0 = *(const uint4v*)pa0;  pa0 += BK;
    ra1 = *(const uint4v*)pa1;  pa1 += BK;
    rb0 = *(const uint4v*)pb;   pb  += BK;
  };
  auto storeTiles = [&](int b) {
    const uint4v ZV = {0u, 0u, 0u, 0u};
    uint4v tb = bOob ? ZV : rb0;                // select at store time
    *(uint4v*)&lA[b][sr        * KPAD + sq] = ra0;
    *(uint4v*)&lA[b][(sr + 64) * KPAD + sq] = ra1;
    *(uint4v*)&lB[b][sr        * KPAD + sq] = tb;
  };

  const int KT = K >> 5;
  loadTiles();
  storeTiles(0);
  int cur = 0;

  for (int kt = 0; kt < KT; ++kt) {
    const bool hasNext = (kt + 1) < KT;         // wave-uniform
    __syncthreads();                            // current buffer ready
    if (hasNext) loadTiles();                   // issue AFTER barrier fence

    // B fragments: lane lo -> col, K = hi*16 .. hi*16+15 contiguous
    FragBF fb[2];
    #pragma unroll
    for (int j = 0; j < 2; ++j) {
      const unsigned short* p = &lB[cur][(wn * 32 + j * 16 + lo) * KPAD + hi * 16];
      fb[j].q[0] = *(const uint4v*)p;
      fb[j].q[1] = *(const uint4v*)(p + 8);
    }
    // A fragments: lane lo -> row, K = hi*16 .. hi*16+15 contiguous
    #pragma unroll
    for (int i = 0; i < 2; ++i) {
      FragBF fa;
      const unsigned short* p = &lA[cur][(wm * 32 + i * 16 + lo) * KPAD + hi * 16];
      fa.q[0] = *(const uint4v*)p;
      fa.q[1] = *(const uint4v*)(p + 8);
      #pragma unroll
      for (int j = 0; j < 2; ++j) {
        acc[i][j] = __builtin_amdgcn_wmma_f32_16x16x32_bf16(
            false, fa.v, false, fb[j].v, (short)0, acc[i][j], false, false);
      }
    }

    if (hasNext) storeTiles(cur ^ 1);           // waits loads here, post-WMMA
    cur ^= 1;
  }

  // Epilogue: D layout: lanes 0-15 -> M=r, lanes 16-31 -> M=r+8, N=lane&15
  #pragma unroll
  for (int i = 0; i < 2; ++i) {
    #pragma unroll
    for (int j = 0; j < 2; ++j) {
      int col = colBase + wn * 32 + j * 16 + lo;
      if (col < N) {
        float bv = bias[col];
        int row0 = rowBase + wm * 32 + i * 16 + hi * 8;
        #pragma unroll
        for (int r = 0; r < 8; ++r) {
          float v = acc[i][j][r] + bv;
          if (RELU) v = fmaxf(v, 0.f);
          size_t o = (size_t)(row0 + r) * N + col;
          if (OUT_BF16) Cb[o] = f32_to_bf16_bits(v);
          else          Cf[o] = v;
        }
      }
    }
  }
}

// ---------------------------------------------------------------------------
extern "C" void kernel_launch(void* const* d_in, const int* in_sizes, int n_in,
                              void* d_out, int out_size, void* d_ws, size_t ws_size,
                              hipStream_t stream) {
  const float* f3     = (const float*)d_in[0];
  const float* f4     = (const float*)d_in[1];
  const float* f5     = (const float*)d_in[2];
  const float* boxes  = (const float*)d_in[3];
  const int*   anchor = (const int*)d_in[4];
  const float* W1 = (const float*)d_in[5];
  const float* b1 = (const float*)d_in[6];
  const float* W2 = (const float*)d_in[7];
  const float* b2 = (const float*)d_in[8];
  const float* Wp = (const float*)d_in[9];
  const float* bp = (const float*)d_in[10];
  (void)in_sizes; (void)n_in; (void)out_size; (void)ws_size;

  char* ws = (char*)d_ws;
  size_t off = 0;
  auto alloc = [&](size_t bytes) -> char* {
    char* p = ws + off;
    off += (bytes + 255) & ~(size_t)255;
    return p;
  };
  unsigned short* pooled = (unsigned short*)alloc((size_t)NROI * FLATD * 2);
  unsigned short* w1b    = (unsigned short*)alloc((size_t)FCD * FLATD * 2);
  unsigned short* w2b    = (unsigned short*)alloc((size_t)FCD * FCD * 2);
  unsigned short* wpb    = (unsigned short*)alloc((size_t)OUTD * FCD * 2);
  unsigned short* h1     = (unsigned short*)alloc((size_t)NROI * FCD * 2);
  unsigned short* h2     = (unsigned short*)alloc((size_t)NROI * FCD * 2);

  // Weight conversion to bf16
  {
    int n = FCD * FLATD;
    f32_to_bf16_kernel<<<(n + 255) / 256, 256, 0, stream>>>(W1, w1b, n);
    n = FCD * FCD;
    f32_to_bf16_kernel<<<(n + 255) / 256, 256, 0, stream>>>(W2, w2b, n);
    n = OUTD * FCD;
    f32_to_bf16_kernel<<<(n + 255) / 256, 256, 0, stream>>>(Wp, wpb, n);
  }

  // ROI align -> bf16 pooled [2048, 12544]
  {
    int total = NROI * FLATD;
    roi_align_kernel<<<(total + 255) / 256, 256, 0, stream>>>(
        f3, f4, f5, boxes, anchor, pooled);
  }

  // MLP head: three bf16 WMMA GEMMs, bias+ReLU fused
  dim3 g1(FCD / BN, NROI / BM);               // (16,16) = 256 blocks
  gemm_bf16_wmma<true, true><<<g1, 256, 0, stream>>>(
      pooled, w1b, b1, nullptr, h1, NROI, FCD, FLATD);
  gemm_bf16_wmma<true, true><<<g1, 256, 0, stream>>>(
      h1, w2b, b2, nullptr, h2, NROI, FCD, FCD);
  dim3 g3((OUTD + BN - 1) / BN, NROI / BM);   // (2,16)
  gemm_bf16_wmma<false, false><<<g3, 256, 0, stream>>>(
      h2, wpb, bp, (float*)d_out, nullptr, NROI, OUTD, FCD);
}
